// GraphConvNet_14293651161727
// MI455X (gfx1250) — matrix operand order repaired and checked by reading, not compiled, not run on hardware
//
#include <hip/hip_runtime.h>
#include <hip/hip_bf16.h>

typedef __attribute__((ext_vector_type(16))) _Float16 v16h;
typedef __attribute__((ext_vector_type(8)))  _Float16 v8h;
typedef __attribute__((ext_vector_type(8)))  float    v8f;

#define NN 100000
#define EE 1600000
#define BB 100

// ---------------------------------------------------------------- utilities
__global__ void zero_f32(float* __restrict__ p, long long n) {
  long long t = blockIdx.x * (long long)blockDim.x + threadIdx.x;
  if (t < n) p[t] = 0.0f;
}

__global__ void gather_emb(const int* __restrict__ idx, const float* __restrict__ emb,
                           float* __restrict__ x, int n, int ed) {
  long long t = blockIdx.x * (long long)blockDim.x + threadIdx.x;
  if (t >= (long long)n * ed) return;
  int node = (int)(t / ed), f = (int)(t % ed);
  x[t] = emb[(long long)idx[node] * ed + f];
}

// msg = x[src]*ew scattered into agg[dst] (segment_sum). float4 per thread.
__global__ void edge_scatter(const int* __restrict__ src, const int* __restrict__ dst,
                             const float* __restrict__ ew, const float* __restrict__ x,
                             float* __restrict__ agg, int e_cnt, int F) {
  int q = F >> 2;
  long long t = blockIdx.x * (long long)blockDim.x + threadIdx.x;
  long long total = (long long)e_cnt * q;
  if (t >= total) return;
  int e = (int)(t / q);
  int f = (int)(t % q) << 2;
  int s = src[e], d = dst[e];
  float w = ew[e];
  const float4 v = *reinterpret_cast<const float4*>(x + (long long)s * F + f);
  float* ap = agg + (long long)d * F + f;
  atomicAdd(ap + 0, v.x * w);
  atomicAdd(ap + 1, v.y * w);
  atomicAdd(ap + 2, v.z * w);
  atomicAdd(ap + 3, v.w * w);
}

// ---------------------------------------------------------------- f16 packing
// A[rowsPad x Kp] = f16([ s1 | s2 | 0-pad ]) row-wise; rows >= rowsReal are zero.
__global__ void pack_a2_f16(const float* __restrict__ s1, int d1,
                            const float* __restrict__ s2, int d2,
                            _Float16* __restrict__ out, int rowsReal, int rowsPad, int Kp) {
  long long t = blockIdx.x * (long long)blockDim.x + threadIdx.x;
  if (t >= (long long)rowsPad * Kp) return;
  int r = (int)(t / Kp), k = (int)(t % Kp);
  float v = 0.0f;
  if (r < rowsReal) {
    if (k < d1) v = s1[(long long)r * d1 + k];
    else if (k < d1 + d2) v = s2[(long long)r * d2 + (k - d1)];
  }
  out[t] = (_Float16)v;
}

// Wt[Mo x Kp] = f16 transpose of [w1; w2] (each row-major K x Mo), zero-padded in k.
__global__ void pack_w2_f16(const float* __restrict__ w1, int d1,
                            const float* __restrict__ w2, int d2,
                            _Float16* __restrict__ out, int Mo, int Kp) {
  int t = blockIdx.x * blockDim.x + threadIdx.x;
  if (t >= Mo * Kp) return;
  int c = t / Kp, k = t % Kp;
  float v = 0.0f;
  if (k < d1) v = w1[k * Mo + c];
  else if (k < d1 + d2) v = w2[(k - d1) * Mo + c];
  out[t] = (_Float16)v;
}

// ---------------------------------------------------------------- WMMA GEMM
// out[rowsReal x Mo] = A[rowsPad x Kp](f16) @ Wt[Mo x Kp]^T(f16) + bias.
// Kp % 32 == 0, rowsPad % 16 == 0, Mo % 64 == 0. One wave -> 16x64 strip.
// All fragment loads are unconditional contiguous b128s.
__global__ void wmma_gemm16(const _Float16* __restrict__ A, const _Float16* __restrict__ Wt,
                            const float* __restrict__ bias, float* __restrict__ out,
                            int rowsPad, int rowsReal, int Mo, int Kp) {
  int wave = threadIdx.x >> 5;
  int lane = threadIdx.x & 31;
  int colGroups = Mo >> 6;
  int totalTiles = (rowsPad >> 4) * colGroups;
  int tile = blockIdx.x * (blockDim.x >> 5) + wave;
  if (tile >= totalTiles) return;               // wave-uniform: EXEC all-ones at WMMA
  int rowTile = tile / colGroups, cg = tile % colGroups;
  int half = lane >> 4, l15 = lane & 15;
  int r  = rowTile * 16 + l15;                  // A row for this lane (always < rowsPad)
  int c0 = cg * 64 + l15;                       // first output column for this lane
  v8f acc[4] = {};
  const _Float16* __restrict__ arow = A + (long long)r * Kp;
  for (int kb = 0; kb < Kp; kb += 32) {
    // A 16x32 f16 layout: e<8 -> k=kb+8*half+e ; e>=8 -> k=kb+16+8*half+(e-8)
    const v8h* ap = reinterpret_cast<const v8h*>(arow + kb + half * 8);
    v8h alo = ap[0];
    v8h ahi = ap[2];                            // +16 halfs
    v16h af = __builtin_shufflevector(alo, ahi, 0, 1, 2, 3, 4, 5, 6, 7,
                                      8, 9, 10, 11, 12, 13, 14, 15);
#pragma unroll
    for (int ct = 0; ct < 4; ++ct) {
      // B 32x16 f16 layout: e<8 -> k=kb+8*half+e ; e>=8 -> k=kb+16+8*half+(e-8)
      const v8h* bp = reinterpret_cast<const v8h*>(Wt + (long long)(c0 + 16 * ct) * Kp + kb + half * 8);
      v8h blo = bp[0];
      v8h bhi = bp[2];
      v16h bf = __builtin_shufflevector(blo, bhi, 0, 1, 2, 3, 4, 5, 6, 7,
                                        8, 9, 10, 11, 12, 13, 14, 15);
      acc[ct] = __builtin_amdgcn_wmma_f32_16x16x32_f16(false, af, false, bf,
                                                       (short)0, acc[ct], false, false);
    }
  }
#pragma unroll
  for (int ct = 0; ct < 4; ++ct) {
    int c = c0 + 16 * ct;
    float bc = bias[c];
#pragma unroll
    for (int v = 0; v < 8; ++v) {
      int m = rowTile * 16 + v + 8 * half;      // C/D layout: VGPR v, lane-half
      if (m < rowsReal) out[(long long)m * Mo + c] = acc[ct][v] + bc;
    }
  }
}

// ---------------------------------------------------------------- segment stats
__global__ void seg_count(const int* __restrict__ batch, float* __restrict__ cnt, int n) {
  int t = blockIdx.x * blockDim.x + threadIdx.x;
  if (t < n) atomicAdd(&cnt[batch[t]], 1.0f);
}

__global__ void seg_sum_feat(const float* __restrict__ x, const int* __restrict__ batch,
                             float* __restrict__ sum, int n, int F) {
  long long t = blockIdx.x * (long long)blockDim.x + threadIdx.x;
  if (t >= (long long)n * F) return;
  int node = (int)(t / F), f = (int)(t % F);
  atomicAdd(&sum[batch[node] * F + f], x[t]);
}

__global__ void mean_from_sum(const float* __restrict__ sum, const float* __restrict__ cnt,
                              float* __restrict__ mean, int bf, int F) {
  int t = blockIdx.x * blockDim.x + threadIdx.x;
  if (t < bf) mean[t] = sum[t] / fmaxf(cnt[t / F], 1.0f);
}

__global__ void div_count(float* __restrict__ io, const float* __restrict__ cnt, int bf, int F) {
  int t = blockIdx.x * blockDim.x + threadIdx.x;
  if (t < bf) io[t] = io[t] / fmaxf(cnt[t / F], 1.0f);
}

__global__ void seg_sumsq_centered(const float* __restrict__ x, const int* __restrict__ batch,
                                   const float* __restrict__ mean, const float* __restrict__ a,
                                   float* __restrict__ vsum, int n, int F) {
  long long t = blockIdx.x * (long long)blockDim.x + threadIdx.x;
  if (t >= (long long)n * F) return;
  int node = (int)(t / F), f = (int)(t % F);
  int b = batch[node];
  float xc = x[t] - a[f] * mean[b * F + f];
  atomicAdd(&vsum[b * F + f], xc * xc);
}

__global__ void gn_apply_lrelu(float* __restrict__ x, const int* __restrict__ batch,
                               const float* __restrict__ mean, const float* __restrict__ var,
                               const float* __restrict__ g, const float* __restrict__ bb,
                               const float* __restrict__ a, int n, int F) {
  long long t = blockIdx.x * (long long)blockDim.x + threadIdx.x;
  if (t >= (long long)n * F) return;
  int node = (int)(t / F), f = (int)(t % F);
  int b = batch[node];
  float xc = x[t] - a[f] * mean[b * F + f];
  float y = g[f] * xc * rsqrtf(var[b * F + f] + 1e-5f) + bb[f];
  x[t] = (y > 0.0f) ? y : 0.01f * y;
}

// ---------------------------------------------------------------- head
__global__ void demo_proj(const float* __restrict__ d, const float* __restrict__ w,
                          const float* __restrict__ bias, float* __restrict__ out, int bn) {
  int t = blockIdx.x * blockDim.x + threadIdx.x;
  if (t >= bn * 16) return;
  int b = t / 16, j = t % 16;
  float s = bias[j];
#pragma unroll
  for (int k = 0; k < 4; ++k) s += d[b * 4 + k] * w[k * 16 + j];
  out[t] = s;
}

__global__ void bn_lrelu(float* __restrict__ h, const float* __restrict__ g,
                         const float* __restrict__ bbias, int bn, int F) {
  int f = threadIdx.x;
  if (f >= F) return;
  float s = 0.0f, s2 = 0.0f;
  for (int i = 0; i < bn; ++i) { float v = h[i * F + f]; s += v; s2 += v * v; }
  float m = s / bn;
  float var = s2 / bn - m * m;
  float inv = rsqrtf(var + 1e-5f);
  float gg = g[f], bb2 = bbias[f];
  for (int i = 0; i < bn; ++i) {
    float v = (h[i * F + f] - m) * inv * gg + bb2;
    h[i * F + f] = (v > 0.0f) ? v : 0.01f * v;
  }
}

__global__ void final_lin(const float* __restrict__ h, const float* __restrict__ w,
                          const float* __restrict__ bias, float* __restrict__ out, int bn, int K) {
  int t = blockIdx.x * blockDim.x + threadIdx.x;
  if (t >= bn * 3) return;
  int b = t / 3, j = t % 3;
  float s = bias[j];
  for (int k = 0; k < K; ++k) s += h[b * K + k] * w[k * 3 + j];
  out[t] = s;
}

// ---------------------------------------------------------------- launch
static inline unsigned gridFor(long long n, int blk) { return (unsigned)((n + blk - 1) / blk); }

extern "C" void kernel_launch(void* const* d_in, const int* in_sizes, int n_in,
                              void* d_out, int out_size, void* d_ws, size_t ws_size,
                              hipStream_t stream) {
  const int*   x_idx   = (const int*)d_in[0];
  const int*   eidx    = (const int*)d_in[1];   // (2,E)
  const float* eattr   = (const float*)d_in[2]; // (E,1)
  const int*   batch   = (const int*)d_in[3];
  const float* demog   = (const float*)d_in[4]; // (B,4)
  const float* emb     = (const float*)d_in[5]; // (V,16)
  const float* w_rel1  = (const float*)d_in[6];
  const float* b_rel1  = (const float*)d_in[7];
  const float* w_root1 = (const float*)d_in[8];
  const float* w_rel2  = (const float*)d_in[9];
  const float* b_rel2  = (const float*)d_in[10];
  const float* w_root2 = (const float*)d_in[11];
  const float* w_rel3  = (const float*)d_in[12];
  const float* b_rel3  = (const float*)d_in[13];
  const float* w_root3 = (const float*)d_in[14];
  const float* gn1_g   = (const float*)d_in[15];
  const float* gn1_b   = (const float*)d_in[16];
  const float* gn1_a   = (const float*)d_in[17];
  const float* gn2_g   = (const float*)d_in[18];
  const float* gn2_b   = (const float*)d_in[19];
  const float* gn2_a   = (const float*)d_in[20];
  const float* demo_w  = (const float*)d_in[21];
  const float* demo_b  = (const float*)d_in[22];
  const float* l1_w    = (const float*)d_in[23]; // (272,128)
  const float* l1_b    = (const float*)d_in[24];
  const float* bn1_g   = (const float*)d_in[25];
  const float* bn1_b   = (const float*)d_in[26];
  const float* l2_w    = (const float*)d_in[27]; // (128,64)
  const float* l2_b    = (const float*)d_in[28];
  const float* bn2_g   = (const float*)d_in[29];
  const float* bn2_b   = (const float*)d_in[30];
  const float* l3_w    = (const float*)d_in[31]; // (64,3)
  const float* l3_b    = (const float*)d_in[32];

  const int* src = eidx;
  const int* dst = eidx + EE;

  float* ws = (float*)d_ws;
  size_t o = 0;
  float* bufX   = ws + o; o += (size_t)NN * 128;   // x0(16) then h2(128)
  float* bufY   = ws + o; o += (size_t)NN * 256;   // h1(64) then h3(256)
  float* bufAgg = ws + o; o += (size_t)NN * 128;
  float* sumB   = ws + o; o += (size_t)BB * 256;
  float* meanB  = ws + o; o += (size_t)BB * 256;
  float* varB   = ws + o; o += (size_t)BB * 256;
  float* cntB   = ws + o; o += 128;                // BB rounded up, keeps alignment
  float* gf     = ws + o; o += (size_t)BB * 256;
  float* demo   = ws + o; o += (size_t)BB * 16;
  float* h1h    = ws + o; o += (size_t)BB * 128;
  float* h2h    = ws + o; o += (size_t)BB * 64;
  _Float16* Af16 = (_Float16*)(ws + o); o += (size_t)NN * 128;  // N x 256 halfs
  _Float16* Wf16 = (_Float16*)(ws + o); o += 48 * 1024;         // up to 96K halfs

  const int BLK = 256;

  // ---- gather node embeddings: x0 = emb[x_idx]  (N x 16)
  gather_emb<<<gridFor((long long)NN * 16, BLK), BLK, 0, stream>>>(x_idx, emb, bufX, NN, 16);

  // ================= conv1: h1 = agg1@w_rel1 + x0@w_root1 + b  (K=16+16 -> Kp=32, Mo=64)
  zero_f32<<<gridFor((long long)NN * 16, BLK), BLK, 0, stream>>>(bufAgg, (long long)NN * 16);
  edge_scatter<<<gridFor((long long)EE * 4, BLK), BLK, 0, stream>>>(src, dst, eattr, bufX, bufAgg, EE, 16);
  pack_a2_f16<<<gridFor((long long)NN * 32, BLK), BLK, 0, stream>>>(bufAgg, 16, bufX, 16, Af16, NN, NN, 32);
  pack_w2_f16<<<gridFor(64 * 32, BLK), BLK, 0, stream>>>(w_rel1, 16, w_root1, 16, Wf16, 64, 32);
  wmma_gemm16<<<gridFor((NN / 16) * (64 / 64), 8), 256, 0, stream>>>(Af16, Wf16, b_rel1, bufY, NN, NN, 64, 32);

  // ---- GN1 + lrelu on bufY (F=64)
  zero_f32<<<gridFor((long long)BB * 256, BLK), BLK, 0, stream>>>(sumB, (long long)BB * 256);
  zero_f32<<<gridFor((long long)BB * 256, BLK), BLK, 0, stream>>>(varB, (long long)BB * 256);
  zero_f32<<<gridFor(BB, BLK), BLK, 0, stream>>>(cntB, BB);
  seg_count<<<gridFor(NN, BLK), BLK, 0, stream>>>(batch, cntB, NN);
  seg_sum_feat<<<gridFor((long long)NN * 64, BLK), BLK, 0, stream>>>(bufY, batch, sumB, NN, 64);
  mean_from_sum<<<gridFor(BB * 64, BLK), BLK, 0, stream>>>(sumB, cntB, meanB, BB * 64, 64);
  seg_sumsq_centered<<<gridFor((long long)NN * 64, BLK), BLK, 0, stream>>>(bufY, batch, meanB, gn1_a, varB, NN, 64);
  div_count<<<gridFor(BB * 64, BLK), BLK, 0, stream>>>(varB, cntB, BB * 64, 64);
  gn_apply_lrelu<<<gridFor((long long)NN * 64, BLK), BLK, 0, stream>>>(bufY, batch, meanB, varB, gn1_g, gn1_b, gn1_a, NN, 64);

  // ================= conv2: h2 = agg2@w_rel2 + h1@w_root2 + b  (Kp=128, Mo=128) -> bufX
  zero_f32<<<gridFor((long long)NN * 64, BLK), BLK, 0, stream>>>(bufAgg, (long long)NN * 64);
  edge_scatter<<<gridFor((long long)EE * 16, BLK), BLK, 0, stream>>>(src, dst, eattr, bufY, bufAgg, EE, 64);
  pack_a2_f16<<<gridFor((long long)NN * 128, BLK), BLK, 0, stream>>>(bufAgg, 64, bufY, 64, Af16, NN, NN, 128);
  pack_w2_f16<<<gridFor(128 * 128, BLK), BLK, 0, stream>>>(w_rel2, 64, w_root2, 64, Wf16, 128, 128);
  wmma_gemm16<<<gridFor((NN / 16) * (128 / 64), 8), 256, 0, stream>>>(Af16, Wf16, b_rel2, bufX, NN, NN, 128, 128);

  // ---- GN2 + lrelu on bufX (F=128)
  zero_f32<<<gridFor((long long)BB * 256, BLK), BLK, 0, stream>>>(sumB, (long long)BB * 256);
  zero_f32<<<gridFor((long long)BB * 256, BLK), BLK, 0, stream>>>(varB, (long long)BB * 256);
  seg_sum_feat<<<gridFor((long long)NN * 128, BLK), BLK, 0, stream>>>(bufX, batch, sumB, NN, 128);
  mean_from_sum<<<gridFor(BB * 128, BLK), BLK, 0, stream>>>(sumB, cntB, meanB, BB * 128, 128);
  seg_sumsq_centered<<<gridFor((long long)NN * 128, BLK), BLK, 0, stream>>>(bufX, batch, meanB, gn2_a, varB, NN, 128);
  div_count<<<gridFor(BB * 128, BLK), BLK, 0, stream>>>(varB, cntB, BB * 128, 128);
  gn_apply_lrelu<<<gridFor((long long)NN * 128, BLK), BLK, 0, stream>>>(bufX, batch, meanB, varB, gn2_g, gn2_b, gn2_a, NN, 128);

  // ================= conv3: h3 = agg3@w_rel3 + h2@w_root3 + b  (Kp=256, Mo=256) -> bufY
  zero_f32<<<gridFor((long long)NN * 128, BLK), BLK, 0, stream>>>(bufAgg, (long long)NN * 128);
  edge_scatter<<<gridFor((long long)EE * 32, BLK), BLK, 0, stream>>>(src, dst, eattr, bufX, bufAgg, EE, 128);
  pack_a2_f16<<<gridFor((long long)NN * 256, BLK), BLK, 0, stream>>>(bufAgg, 128, bufX, 128, Af16, NN, NN, 256);
  pack_w2_f16<<<gridFor(256 * 256, BLK), BLK, 0, stream>>>(w_rel3, 128, w_root3, 128, Wf16, 256, 256);
  wmma_gemm16<<<gridFor((NN / 16) * (256 / 64), 8), 256, 0, stream>>>(Af16, Wf16, b_rel3, bufY, NN, NN, 256, 256);

  // ---- segment-mean pooling: gf = seg_mean(h3)  (B x 256)
  zero_f32<<<gridFor((long long)BB * 256, BLK), BLK, 0, stream>>>(gf, (long long)BB * 256);
  seg_sum_feat<<<gridFor((long long)NN * 256, BLK), BLK, 0, stream>>>(bufY, batch, gf, NN, 256);
  div_count<<<gridFor(BB * 256, BLK), BLK, 0, stream>>>(gf, cntB, BB * 256, 256);

  // ================= head
  demo_proj<<<gridFor(BB * 16, BLK), BLK, 0, stream>>>(demog, demo_w, demo_b, demo, BB);
  // l1: h = [gf | demo] @ l1_w + l1_b   (Kp = 272 -> 288, Mo = 128, rows 100 -> pad 112)
  pack_a2_f16<<<gridFor((long long)112 * 288, BLK), BLK, 0, stream>>>(gf, 256, demo, 16, Af16, BB, 112, 288);
  pack_w2_f16<<<gridFor(128 * 288, BLK), BLK, 0, stream>>>(l1_w, 272, nullptr, 0, Wf16, 128, 288);
  wmma_gemm16<<<gridFor((112 / 16) * (128 / 64), 8), 256, 0, stream>>>(Af16, Wf16, l1_b, h1h, 112, BB, 128, 288);
  bn_lrelu<<<1, 128, 0, stream>>>(h1h, bn1_g, bn1_b, BB, 128);
  // l2: (Kp = 128, Mo = 64)
  pack_a2_f16<<<gridFor((long long)112 * 128, BLK), BLK, 0, stream>>>(h1h, 128, nullptr, 0, Af16, BB, 112, 128);
  pack_w2_f16<<<gridFor(64 * 128, BLK), BLK, 0, stream>>>(l2_w, 128, nullptr, 0, Wf16, 64, 128);
  wmma_gemm16<<<gridFor((112 / 16) * (64 / 64), 8), 256, 0, stream>>>(Af16, Wf16, l2_b, h2h, 112, BB, 64, 128);
  bn_lrelu<<<1, 64, 0, stream>>>(h2h, bn2_g, bn2_b, BB, 64);
  final_lin<<<gridFor(BB * 3, BLK), BLK, 0, stream>>>(h2h, l3_w, l3_b, (float*)d_out, BB, 64);
}
